// SiLUAttention_29635274343033
// MI455X (gfx1250) — compile-verified, matrix operand order checked by reading, not run on hardware
//
#include <hip/hip_runtime.h>
#include <hip/hip_bf16.h>

// ---------------------------------------------------------------------------
// SiLU-attention for MI455X (gfx1250, wave32).
// All GEMM stages on v_wmma_f32_16x16x32_bf16, fp32 accumulation.
// Weights pre-transposed+converted to bf16 so every WMMA fragment is two
// contiguous global_load_b128 per lane (ISA 7.12.2 16-bit layout).
// ---------------------------------------------------------------------------

typedef __bf16 bf16;
typedef __attribute__((ext_vector_type(16))) __bf16 v16bf;
typedef __attribute__((ext_vector_type(8)))  __bf16 v8bf;
typedef __attribute__((ext_vector_type(8)))  float  v8f;

#define WMMA_BF16(a, b, c)                                                \
    __builtin_amdgcn_wmma_f32_16x16x32_bf16(false, (a), false, (b),       \
                                            (short)0, (c), false, false)

#define BATCH 2
#define SEQ   2048
#define DIM   1024
#define HEADS 16
#define HD    64
#define N4    4096   // 4*DIM
#define MROWS 4096   // BATCH*SEQ

__device__ __forceinline__ float fast_silu(float x) {
    return x * __builtin_amdgcn_rcpf(1.0f + __expf(-x));
}

// Load a 16x32 bf16 fragment row-chunk for this lane: 8 bf16 at p, 8 at p+16.
// Caller passes p already offset by (k0 + 8*half).
__device__ __forceinline__ v16bf ld_frag(const bf16* p) {
    v8bf lo = *(const v8bf*)(p);
    v8bf hi = *(const v8bf*)(p + 16);
    return __builtin_shufflevector(lo, hi, 0, 1, 2, 3, 4, 5, 6, 7,
                                           8, 9, 10, 11, 12, 13, 14, 15);
}

// ---------------------------------------------------------------------------
// Prep kernel A: fp32 -> bf16 bulk convert (8 elements / thread).
// ---------------------------------------------------------------------------
__global__ __launch_bounds__(256)
void cvt_f32_bf16(const float* __restrict__ src, bf16* __restrict__ dst, int n) {
    const int i = (blockIdx.x * 256 + threadIdx.x) * 8;
    if (i >= n) return;
    float4 a = *(const float4*)(src + i);
    float4 b = *(const float4*)(src + i + 4);
    v8bf o;
    o[0] = (bf16)a.x; o[1] = (bf16)a.y; o[2] = (bf16)a.z; o[3] = (bf16)a.w;
    o[4] = (bf16)b.x; o[5] = (bf16)b.y; o[6] = (bf16)b.z; o[7] = (bf16)b.w;
    *(v8bf*)(dst + i) = o;
}

// ---------------------------------------------------------------------------
// Prep kernel B: transpose + convert, dst[n][k] = (bf16)src[k][n].
// 32x32 LDS tile, coalesced both directions.
// ---------------------------------------------------------------------------
__global__ __launch_bounds__(256)
void transpose_cvt(const float* __restrict__ src, bf16* __restrict__ dst,
                   int K, int N) {
    __shared__ bf16 t[32][33];                 // +1 pad kills bank conflicts
    const int tx = threadIdx.x & 31;
    const int ty = threadIdx.x >> 5;           // 0..7
    const int bn = blockIdx.x % (N / 32);
    const int bk = blockIdx.x / (N / 32);
    const int n0 = bn * 32, k0 = bk * 32;
#pragma unroll
    for (int r = 0; r < 4; ++r)
        t[ty + 8 * r][tx] = (bf16)src[(size_t)(k0 + ty + 8 * r) * N + n0 + tx];
    __syncthreads();
#pragma unroll
    for (int r = 0; r < 4; ++r)
        dst[(size_t)(n0 + ty + 8 * r) * K + k0 + tx] = t[tx][ty + 8 * r];
}

// ---------------------------------------------------------------------------
// Kernel 1: QKUV = silu(Xbf @ Wqkuv). 32(M) x 64(N) per wave, 8 WMMA / K-step.
// V head-slices are written TRANSPOSED into VT[b,h,hd,key] so the attention
// w@v B-operand later loads contiguously.
// ---------------------------------------------------------------------------
__global__ __launch_bounds__(256)
void qkuv_gemm_silu(const bf16* __restrict__ X,    // [4096][1024]
                    const bf16* __restrict__ WT,   // [4096][1024] (N-major)
                    bf16* __restrict__ QKUV,       // [4096][4096]
                    bf16* __restrict__ VT) {       // [B*H*HD][SEQ]
    const int lane = threadIdx.x & 31;
    const int wave = threadIdx.x >> 5;
    const int half = lane >> 4;
    const int ln   = lane & 15;

    const int tiles_n = N4 / 64;                   // 64
    const int tile    = blockIdx.x * 8 + wave;     // 8192 tiles exactly
    const int m0      = (tile / tiles_n) * 32;
    const int n0      = (tile % tiles_n) * 64;

    const bf16* ar0 = X + (size_t)(m0 + ln) * DIM + 8 * half;
    const bf16* ar1 = X + (size_t)(m0 + 16 + ln) * DIM + 8 * half;

    v8f acc[2][4] = {};
    for (int k0 = 0; k0 < DIM; k0 += 32) {
        __builtin_prefetch(ar0 + k0 + 128, 0, 1);
        __builtin_prefetch(ar1 + k0 + 128, 0, 1);
        v16bf a0 = ld_frag(ar0 + k0);
        v16bf a1 = ld_frag(ar1 + k0);
#pragma unroll
        for (int nt = 0; nt < 4; ++nt) {
            const bf16* bp = WT + (size_t)(n0 + nt * 16 + ln) * DIM + 8 * half;
            v16bf bfr = ld_frag(bp + k0);
            acc[0][nt] = WMMA_BF16(a0, bfr, acc[0][nt]);
            acc[1][nt] = WMMA_BF16(a1, bfr, acc[1][nt]);
        }
    }

    const bool is_v = (n0 >= 2 * DIM) && (n0 < 3 * DIM);
    if (is_v) {
        const int h     = (n0 - 2 * DIM) >> 6;
        const int b     = m0 / SEQ;
        const int sbase = m0 % SEQ;
        bf16* vbase = VT + (size_t)(b * HEADS + h) * HD * SEQ;
#pragma unroll
        for (int mi = 0; mi < 2; ++mi)
#pragma unroll
            for (int nt = 0; nt < 4; ++nt)
#pragma unroll
                for (int i = 0; i < 8; ++i) {
                    const int s  = sbase + mi * 16 + i + 8 * half;
                    const int hd = nt * 16 + ln;
                    vbase[(size_t)hd * SEQ + s] = (bf16)fast_silu(acc[mi][nt][i]);
                }
    } else {
#pragma unroll
        for (int mi = 0; mi < 2; ++mi)
#pragma unroll
            for (int nt = 0; nt < 4; ++nt)
#pragma unroll
                for (int i = 0; i < 8; ++i) {
                    const int row = m0 + mi * 16 + i + 8 * half;
                    QKUV[(size_t)row * N4 + n0 + nt * 16 + ln] =
                        (bf16)fast_silu(acc[mi][nt][i]);
                }
    }
}

// ---------------------------------------------------------------------------
// Kernel 2: streaming SiLU attention. One wave per 32-query block per (b,h):
// k/v fragments are loaded once per 32-key step and reused across both
// 16-query row groups (16 WMMA per 16 b128 loads).
// scores: 2 WMMA over hd=64 per (sub, mi); w re-laid out C->A via per-wave
// LDS tiles; w@v: 8 WMMA per 32 keys with VT giving contiguous B loads.
// ---------------------------------------------------------------------------
__global__ __launch_bounds__(256)
void attn_kernel(const bf16* __restrict__ QKUV,
                 const bf16* __restrict__ VT,
                 const unsigned char* __restrict__ mask,
                 const float* __restrict__ gamma,
                 const float* __restrict__ beta,
                 bf16* __restrict__ ATTG) {
    __shared__ bf16 wtile[8][2][16][32];           // 16 KB, two tiles per wave

    const int lane = threadIdx.x & 31;
    const int wave = threadIdx.x >> 5;
    const int half = lane >> 4;
    const int ln   = lane & 15;

    const int qblocks = SEQ / 32;                  // 64
    const int tile = blockIdx.x * 8 + wave;        // 2048 tiles exactly
    const int b  = tile / (HEADS * qblocks);
    const int h  = (tile / qblocks) % HEADS;
    const int qb = tile % qblocks;

    // Q fragments for two 16-row groups, hd=64 split in two K-chunks.
    v16bf qf[2][2];
#pragma unroll
    for (int mi = 0; mi < 2; ++mi) {
        const bf16* qp = QKUV + (size_t)(b * SEQ + qb * 32 + mi * 16 + ln) * N4 +
                         (size_t)h * HD + 8 * half;
        qf[mi][0] = ld_frag(qp);
        qf[mi][1] = ld_frag(qp + 32);
    }

    const bf16* vbase = VT + (size_t)(b * HEADS + h) * HD * SEQ;

    float dsum[2][8];
#pragma unroll
    for (int mi = 0; mi < 2; ++mi)
#pragma unroll
        for (int i = 0; i < 8; ++i) dsum[mi][i] = 0.0f;
    v8f att[2][4] = {};

    for (int kb = 0; kb < SEQ; kb += 32) {
        // ---- scores for two 16-key subtiles x two query groups -----------
#pragma unroll
        for (int sub = 0; sub < 2; ++sub) {
            const int kbs = kb + sub * 16;
            const bf16* kp = QKUV + (size_t)(b * SEQ + kbs + ln) * N4 + DIM +
                             (size_t)h * HD + 8 * half;
            const v16bf kf0 = ld_frag(kp);
            const v16bf kf1 = ld_frag(kp + 32);
#pragma unroll
            for (int mi = 0; mi < 2; ++mi) {
                v8f sc = {};
                sc = WMMA_BF16(qf[mi][0], kf0, sc);
                sc = WMMA_BF16(qf[mi][1], kf1, sc);
#pragma unroll
                for (int i = 0; i < 8; ++i) {
                    const int qrow = qb * 32 + mi * 16 + i + 8 * half;
                    const int key  = kbs + ln;
                    float w = fast_silu(sc[i] * 0.125f);   // 1/sqrt(64)
                    w *= (float)mask[(size_t)qrow * SEQ + key];
                    dsum[mi][i] += fabsf(w);
                    wtile[wave][mi][i + 8 * half][sub * 16 + ln] = (bf16)w;
                }
            }
        }
        __syncthreads();

        // ---- w (A frags from LDS) x v (B frags from VT): 8 WMMAs ---------
        v16bf wf0 = ld_frag(&wtile[wave][0][ln][8 * half]);
        v16bf wf1 = ld_frag(&wtile[wave][1][ln][8 * half]);
#pragma unroll
        for (int nt = 0; nt < 4; ++nt) {
            v16bf vf = ld_frag(vbase + (size_t)(nt * 16 + ln) * SEQ + kb + 8 * half);
            att[0][nt] = WMMA_BF16(wf0, vf, att[0][nt]);
            att[1][nt] = WMMA_BF16(wf1, vf, att[1][nt]);
        }
        __syncthreads();
    }

#pragma unroll
    for (int mi = 0; mi < 2; ++mi) {
        // clip(sum|w|, 1, inf) across the 16-lane half-group, then scale.
#pragma unroll
        for (int i = 0; i < 8; ++i) {
            float d = dsum[mi][i];
            d += __shfl_xor(d, 1, 32);
            d += __shfl_xor(d, 2, 32);
            d += __shfl_xor(d, 4, 32);
            d += __shfl_xor(d, 8, 32);
            const float inv = __builtin_amdgcn_rcpf(fmaxf(d, 1.0f));
#pragma unroll
            for (int nt = 0; nt < 4; ++nt) att[mi][nt][i] *= inv;
        }

        // Per-row LayerNorm over hd=64, gate with g, store bf16.
#pragma unroll
        for (int i = 0; i < 8; ++i) {
            float s = 0.0f, sq = 0.0f;
#pragma unroll
            for (int nt = 0; nt < 4; ++nt) {
                const float x = att[mi][nt][i];
                s += x; sq += x * x;
            }
            s  += __shfl_xor(s, 1, 32);  s  += __shfl_xor(s, 2, 32);
            s  += __shfl_xor(s, 4, 32);  s  += __shfl_xor(s, 8, 32);
            sq += __shfl_xor(sq, 1, 32); sq += __shfl_xor(sq, 2, 32);
            sq += __shfl_xor(sq, 4, 32); sq += __shfl_xor(sq, 8, 32);
            const float mu   = s * (1.0f / (float)HD);
            const float var  = sq * (1.0f / (float)HD) - mu * mu;
            const float rstd = __builtin_amdgcn_rsqf(var + 1e-5f);

            const int    mrow = b * SEQ + qb * 32 + mi * 16 + i + 8 * half;
            const size_t grow = (size_t)mrow * N4 + 3 * DIM + (size_t)h * HD;
#pragma unroll
            for (int nt = 0; nt < 4; ++nt) {
                const int col = nt * 16 + ln;
                float x = (att[mi][nt][i] - mu) * rstd * gamma[col] + beta[col];
                x *= (float)QKUV[grow + col];              // g gate
                ATTG[(size_t)mrow * DIM + (size_t)h * HD + col] = (bf16)x;
            }
        }
    }
}

// ---------------------------------------------------------------------------
// Kernel 3: OUT = ATTG(bf16) @ Wout. 32x64 per wave, fp32 output.
// ---------------------------------------------------------------------------
__global__ __launch_bounds__(256)
void out_gemm(const bf16* __restrict__ A,          // [4096][1024]
              const bf16* __restrict__ WT,         // [1024][1024] (N-major)
              float* __restrict__ OUT) {
    const int lane = threadIdx.x & 31;
    const int wave = threadIdx.x >> 5;
    const int half = lane >> 4;
    const int ln   = lane & 15;

    const int tiles_n = DIM / 64;                  // 16
    const int tile    = blockIdx.x * 8 + wave;     // 2048 tiles exactly
    const int m0      = (tile / tiles_n) * 32;
    const int n0      = (tile % tiles_n) * 64;

    const bf16* ar0 = A + (size_t)(m0 + ln) * DIM + 8 * half;
    const bf16* ar1 = A + (size_t)(m0 + 16 + ln) * DIM + 8 * half;

    v8f acc[2][4] = {};
    for (int k0 = 0; k0 < DIM; k0 += 32) {
        __builtin_prefetch(ar0 + k0 + 128, 0, 1);
        v16bf a0 = ld_frag(ar0 + k0);
        v16bf a1 = ld_frag(ar1 + k0);
#pragma unroll
        for (int nt = 0; nt < 4; ++nt) {
            const bf16* bp = WT + (size_t)(n0 + nt * 16 + ln) * DIM + 8 * half;
            v16bf bfr = ld_frag(bp + k0);
            acc[0][nt] = WMMA_BF16(a0, bfr, acc[0][nt]);
            acc[1][nt] = WMMA_BF16(a1, bfr, acc[1][nt]);
        }
    }
#pragma unroll
    for (int mi = 0; mi < 2; ++mi)
#pragma unroll
        for (int nt = 0; nt < 4; ++nt)
#pragma unroll
            for (int i = 0; i < 8; ++i) {
                const int row = m0 + mi * 16 + i + 8 * half;
                OUT[(size_t)row * DIM + n0 + nt * 16 + ln] = acc[mi][nt][i];
            }
}

// ---------------------------------------------------------------------------
extern "C" void kernel_launch(void* const* d_in, const int* in_sizes, int n_in,
                              void* d_out, int out_size, void* d_ws, size_t ws_size,
                              hipStream_t stream) {
    const float*         hidden = (const float*)d_in[0];
    const unsigned char* mask   = (const unsigned char*)d_in[1];  // bool[S,S]
    const float*         Wqkuv  = (const float*)d_in[2];
    const float*         Wout   = (const float*)d_in[3];
    const float*         gamma  = (const float*)d_in[4];
    const float*         beta   = (const float*)d_in[5];
    float*               out    = (float*)d_out;

    char* ws = (char*)d_ws;
    bf16* Xbf    = (bf16*)(ws);                               //  8 MB
    bf16* WqkuvT = (bf16*)(ws + (8u << 20));                  //  8 MB
    bf16* WoutT  = (bf16*)(ws + (16u << 20));                 //  2 MB
    bf16* QKUV   = (bf16*)(ws + (18u << 20));                 // 32 MB
    bf16* VT     = (bf16*)(ws + (50u << 20));                 //  8 MB
    bf16* ATTG   = (bf16*)(ws + (58u << 20));                 //  8 MB

    // Prep: bf16 convert X, transpose+convert weights.
    cvt_f32_bf16<<<(MROWS * DIM) / (256 * 8), 256, 0, stream>>>(hidden, Xbf,
                                                                MROWS * DIM);
    transpose_cvt<<<(N4 / 32) * (DIM / 32), 256, 0, stream>>>(Wqkuv, WqkuvT,
                                                              DIM, N4);
    transpose_cvt<<<(DIM / 32) * (DIM / 32), 256, 0, stream>>>(Wout, WoutT,
                                                               DIM, DIM);

    // 1) QKUV = silu(X @ Wqkuv): (4096/32)*(4096/64) = 8192 wave tiles.
    qkuv_gemm_silu<<<1024, 256, 0, stream>>>(Xbf, WqkuvT, QKUV, VT);

    // 2) attention: B*H*(S/32) = 2048 wave tiles.
    attn_kernel<<<256, 256, 0, stream>>>(QKUV, VT, mask, gamma, beta, ATTG);

    // 3) OUT = attg @ Wout: (4096/32)*(1024/64) = 2048 wave tiles.
    out_gemm<<<256, 256, 0, stream>>>(ATTG, WoutT, out);
}
